// SimpleRNN_41171556500011
// MI455X (gfx1250) — matrix operand order repaired and checked by reading, not compiled, not run on hardware
//
#include <hip/hip_runtime.h>
#include <math.h>

#define VOCAB  128
#define HIDDEN 1024
#define OUTDIM 128
#define BATCH  64
#define SEQ    512
#define NWG    16     // one WG per 64 hidden columns
#define NTHR   256    // 8 waves of 32

// ---- workspace layout (bytes) ----
#define WS_CTR 0
#define WS_H0  1024
#define WS_H1  (WS_H0 + BATCH*HIDDEN*2)                 // bf16 h ping-pong buffers
#define WS_WHH (WS_H1 + BATCH*HIDDEN*2)                 // 2048 frags * 1024B (bf16)
#define WS_WHY (WS_WHH + 64*32*1024)                    // 256 frags * 1024B (bf16)

// ---- dynamic LDS layout (bytes) ----
#define LDS_WHH 0                        // WG's W_hh slice: 4 nt * 32 kt * 1024B = 128KB
#define LDS_WHY 131072                   // WG's W_hy slice: 32 kt * 1024B = 32KB
#define LDS_XW  163840                   // W_xh slice + b_h: 128*64 f32 = 32KB
#define LDS_BY  196608                   // b_y: 128 f32
#define LDS_TOK 197120                   // tokens: 64 int
#define LDS_TOT 197632

typedef __attribute__((ext_vector_type(16))) __bf16 bf16x16;
typedef __attribute__((ext_vector_type(8)))  float  f32x8;

union Frag { bf16x16 v; uint4 q[2]; };

static __device__ __forceinline__ unsigned short f2bf(float f) {
    unsigned u = __float_as_uint(f);
    u += 0x7FFFu + ((u >> 16) & 1u);      // round to nearest even
    return (unsigned short)(u >> 16);
}

// Saturating fast tanh: exact at +-1 limits, ~3 VALU ops.
static __device__ __forceinline__ float fast_tanh(float x) {
    float e = __builtin_amdgcn_exp2f(2.885390082f * x);   // exp(2x)
    return 1.0f - 2.0f * __builtin_amdgcn_rcpf(e + 1.0f);
}

// Zero barrier counter and both h ping-pong buffers (h_{-1} = 0).
__global__ void rnn_init(unsigned* wsu) {
    int tid = blockIdx.x * NTHR + threadIdx.x;
    if (tid == 0) wsu[WS_CTR / 4] = 0u;
    unsigned* h = wsu + WS_H0 / 4;        // 2 * 64*1024 bf16 = 65536 dwords
    if (tid < 65536) h[tid] = 0u;
}

// W_hh (f32, [K=1024, N=1024] row-major) -> bf16 WMMA B-fragments.
// frag id = nt*32 + kt (nt: 16-col tile, kt: 32-row K tile), 1024B each:
// lane L holds 16 bf16: col = nt*16 + (L&15),
// K = kt*32 + ((L&16)?16:0) + {0..15}, packed 2-per-dword (contiguous K order, same as the A loads).
__global__ void prep_whh(const float* __restrict__ whh, unsigned short* __restrict__ frags) {
    int tid  = blockIdx.x * NTHR + threadIdx.x;     // 65536 = 2048 frags * 32 lanes
    int frag = tid >> 5, lane = tid & 31;
    int nt = frag >> 5, kt = frag & 31;
    int col   = nt * 16 + (lane & 15);
    int kbase = kt * 32 + ((lane & 16) ? 16 : 0);
    unsigned short* dst = frags + frag * 512 + lane * 16;
#pragma unroll
    for (int v = 0; v < 8; ++v) {
        dst[2*v]   = f2bf(whh[(kbase + 2*v    ) * HIDDEN + col]);
        dst[2*v+1] = f2bf(whh[(kbase + 2*v + 1) * HIDDEN + col]);
    }
}

// W_hy (f32, [K=1024, N=128] row-major) -> bf16 B-fragments, same scheme.
__global__ void prep_why(const float* __restrict__ why, unsigned short* __restrict__ frags) {
    int tid  = blockIdx.x * NTHR + threadIdx.x;     // 8192 = 256 frags * 32 lanes
    int frag = tid >> 5, lane = tid & 31;
    int nt = frag >> 5, kt = frag & 31;
    int col   = nt * 16 + (lane & 15);
    int kbase = kt * 32 + ((lane & 16) ? 16 : 0);
    unsigned short* dst = frags + frag * 512 + lane * 16;
#pragma unroll
    for (int v = 0; v < 8; ++v) {
        dst[2*v]   = f2bf(why[(kbase + 2*v    ) * OUTDIM + col]);
        dst[2*v+1] = f2bf(why[(kbase + 2*v + 1) * OUTDIM + col]);
    }
}

// Persistent RNN: 16 WGs, each owns 64 hidden columns; W_hh/W_hy slices live
// in LDS, h ping-pongs through L2. Per step t:
//   acc = h_{t-1} @ W_hh[:, slice]                     (WMMA bf16, K=1024)
//   h_t[:, slice] = tanh(acc + W_xh[token] + b_h)      (bf16 publish)
//   y_{t-1}[:, 16-col slice] = h_{t-1} @ W_hy + b_y    (WGs 0..7, waves 0..3)
// then a 16-party global barrier (atomic counter + agent fences).
__global__ void __launch_bounds__(NTHR)
rnn_persist(const int* __restrict__ X, const float* __restrict__ Wxh,
            const float* __restrict__ bh, const float* __restrict__ by,
            float* __restrict__ out, unsigned char* __restrict__ ws) {
    extern __shared__ char smem[];
    unsigned short* s_whh = (unsigned short*)(smem + LDS_WHH);
    unsigned short* s_why = (unsigned short*)(smem + LDS_WHY);
    float*          s_xw  = (float*)(smem + LDS_XW);
    float*          s_by  = (float*)(smem + LDS_BY);
    int*            s_tok = (int*)(smem + LDS_TOK);

    const int wg   = blockIdx.x;         // 0..15
    const int tid  = threadIdx.x;
    const int lane = tid & 31;
    const int wave = tid >> 5;

    unsigned*       ctr = (unsigned*)(ws + WS_CTR);
    unsigned short* h0  = (unsigned short*)(ws + WS_H0);
    unsigned short* h1  = (unsigned short*)(ws + WS_H1);

    // ---- stage W_hh / W_hy fragment slices into LDS (once; amortized x512) ----
    {
        const uint4* src = (const uint4*)(ws + WS_WHH + (size_t)wg * 131072);
        uint4*       dst = (uint4*)(smem + LDS_WHH);
        for (int i = tid; i < 8192; i += NTHR) dst[i] = src[i];
        if (wg < 8) {
            const uint4* ysrc = (const uint4*)(ws + WS_WHY + (size_t)wg * 32768);
            uint4*       ydst = (uint4*)(smem + LDS_WHY);
            for (int i = tid; i < 2048; i += NTHR) ydst[i] = ysrc[i];
        }
        for (int i = tid; i < VOCAB * 64; i += NTHR) {
            int v = i >> 6, c = i & 63;
            s_xw[i] = Wxh[v * HIDDEN + wg * 64 + c] + bh[wg * 64 + c];
        }
        for (int i = tid; i < OUTDIM; i += NTHR) s_by[i] = by[i];
    }
    __syncthreads();

    const int  m       = wave & 3;              // wave's M (batch) tile 0..3
    const int  n0      = (wave >> 2) * 2;       // wave's local N tiles n0, n0+1
    const bool doYwave = (wave < 4) && (wg < 8);

    const int arow  = m * 16 + (lane & 15);     // batch row for A operand
    const int akoff = (lane & 16) ? 16 : 0;     // A K-chunk select
    const unsigned short* bf0 = s_whh + (n0    ) * 32 * 512 + lane * 16;
    const unsigned short* bf1 = s_whh + (n0 + 1) * 32 * 512 + lane * 16;
    const unsigned short* yf  = s_why + lane * 16;

    for (int t = 0; t <= SEQ; ++t) {
        const bool doRec = (t < SEQ);
        const bool doY   = (t >= 1) && doYwave;
        const unsigned short* hprev = (t & 1) ? h0 : h1;    // h_{t-1}
        unsigned short*       hnext = (t & 1) ? h1 : h0;    // h_t

        if (doRec && tid < BATCH) s_tok[tid] = X[tid * SEQ + t];
        __syncthreads();

        f32x8 acc0 = {}, acc1 = {}, yacc = {};
        const unsigned short* ap = hprev + arow * HIDDEN + akoff;

        // ---- branch-free k-loop variants (conditions are loop-invariant) ----
        if (doRec & doY) {
#pragma unroll 4
            for (int kt = 0; kt < 32; ++kt) {
                Frag a, b0, b1, yb;
                a.q[0]  = *(const uint4*)(ap + kt * 32);
                a.q[1]  = *(const uint4*)(ap + kt * 32 + 8);
                b0.q[0] = *(const uint4*)(bf0 + kt * 512);
                b0.q[1] = *(const uint4*)(bf0 + kt * 512 + 8);
                b1.q[0] = *(const uint4*)(bf1 + kt * 512);
                b1.q[1] = *(const uint4*)(bf1 + kt * 512 + 8);
                yb.q[0] = *(const uint4*)(yf + kt * 512);
                yb.q[1] = *(const uint4*)(yf + kt * 512 + 8);
                acc0 = __builtin_amdgcn_wmma_f32_16x16x32_bf16(
                           false, a.v, false, b0.v, (short)0, acc0, false, false);
                acc1 = __builtin_amdgcn_wmma_f32_16x16x32_bf16(
                           false, a.v, false, b1.v, (short)0, acc1, false, false);
                yacc = __builtin_amdgcn_wmma_f32_16x16x32_bf16(
                           false, a.v, false, yb.v, (short)0, yacc, false, false);
            }
        } else if (doRec) {
#pragma unroll 4
            for (int kt = 0; kt < 32; ++kt) {
                Frag a, b0, b1;
                a.q[0]  = *(const uint4*)(ap + kt * 32);
                a.q[1]  = *(const uint4*)(ap + kt * 32 + 8);
                b0.q[0] = *(const uint4*)(bf0 + kt * 512);
                b0.q[1] = *(const uint4*)(bf0 + kt * 512 + 8);
                b1.q[0] = *(const uint4*)(bf1 + kt * 512);
                b1.q[1] = *(const uint4*)(bf1 + kt * 512 + 8);
                acc0 = __builtin_amdgcn_wmma_f32_16x16x32_bf16(
                           false, a.v, false, b0.v, (short)0, acc0, false, false);
                acc1 = __builtin_amdgcn_wmma_f32_16x16x32_bf16(
                           false, a.v, false, b1.v, (short)0, acc1, false, false);
            }
        } else if (doY) {   // t == SEQ: final y only
#pragma unroll 4
            for (int kt = 0; kt < 32; ++kt) {
                Frag a, yb;
                a.q[0]  = *(const uint4*)(ap + kt * 32);
                a.q[1]  = *(const uint4*)(ap + kt * 32 + 8);
                yb.q[0] = *(const uint4*)(yf + kt * 512);
                yb.q[1] = *(const uint4*)(yf + kt * 512 + 8);
                yacc = __builtin_amdgcn_wmma_f32_16x16x32_bf16(
                           false, a.v, false, yb.v, (short)0, yacc, false, false);
            }
        }

        // ---- recurrence epilogue: x-proj gather, tanh, publish h_t (bf16) ----
        if (doRec) {
            int nloc0 = n0 * 16 + (lane & 15);          // local col within 64
            int rowb  = m * 16 + ((lane & 16) ? 8 : 0);
#pragma unroll
            for (int v = 0; v < 8; ++v) {
                int   row = rowb + v;
                int   tok = s_tok[row];
                float p0  = acc0[v] + s_xw[tok * 64 + nloc0];
                float p1  = acc1[v] + s_xw[tok * 64 + nloc0 + 16];
                hnext[row * HIDDEN + wg * 64 + nloc0]      = f2bf(fast_tanh(p0));
                hnext[row * HIDDEN + wg * 64 + nloc0 + 16] = f2bf(fast_tanh(p1));
            }
        }

        // ---- y_{t-1} epilogue: D tile (m, o-tile wg); C layout N=lane&15, M=v(+8) ----
        if (doY) {
            int o     = wg * 16 + (lane & 15);
            int bbase = m * 16 + ((lane & 16) ? 8 : 0);
            float bias = s_by[o];
#pragma unroll
            for (int v = 0; v < 8; ++v)
                out[(size_t)(bbase + v) * (SEQ * OUTDIM) + (size_t)(t - 1) * OUTDIM + o] =
                    yacc[v] + bias;
        }

        // ---- 16-party global barrier: release stores, arrive, spin, acquire ----
        if (doRec) {
            __builtin_amdgcn_fence(__ATOMIC_RELEASE, "agent");
            __syncthreads();
            if (tid == 0) {
                __hip_atomic_fetch_add(ctr, 1u, __ATOMIC_RELEASE, __HIP_MEMORY_SCOPE_AGENT);
                unsigned target = (unsigned)(NWG * (t + 1));
                while (__hip_atomic_load(ctr, __ATOMIC_ACQUIRE, __HIP_MEMORY_SCOPE_AGENT)
                       < target)
                    __builtin_amdgcn_s_sleep(1);
            }
            __syncthreads();
            __builtin_amdgcn_fence(__ATOMIC_ACQUIRE, "agent");
        }
    }
}

extern "C" void kernel_launch(void* const* d_in, const int* in_sizes, int n_in,
                              void* d_out, int out_size, void* d_ws, size_t ws_size,
                              hipStream_t stream) {
    const int*   X   = (const int*)d_in[0];
    const float* Whh = (const float*)d_in[1];
    const float* Wxh = (const float*)d_in[2];
    const float* Why = (const float*)d_in[3];
    const float* bh  = (const float*)d_in[4];
    const float* by  = (const float*)d_in[5];
    float*         out = (float*)d_out;
    unsigned char* ws  = (unsigned char*)d_ws;

    // Allow >64KB dynamic LDS (WGP supports up to 320KB per workgroup).
    (void)hipFuncSetAttribute((const void*)rnn_persist,
                              hipFuncAttributeMaxDynamicSharedMemorySize, LDS_TOT);

    rnn_init<<<256, NTHR, 0, stream>>>((unsigned*)ws);
    prep_whh<<<256, NTHR, 0, stream>>>(Whh, (unsigned short*)(ws + WS_WHH));
    prep_why<<<32,  NTHR, 0, stream>>>(Why, (unsigned short*)(ws + WS_WHY));
    rnn_persist<<<NWG, NTHR, LDS_TOT, stream>>>(X, Wxh, bh, by, out, ws);
}
